// LSTMTagger_57681410785944
// MI455X (gfx1250) — matrix operand order repaired and checked by reading, not compiled, not run on hardware
//
#include <hip/hip_runtime.h>

// ---------------- types ----------------
typedef __bf16 bf16_t;
typedef __attribute__((ext_vector_type(16))) __bf16 v16bf;
typedef __attribute__((ext_vector_type(8)))  float  v8f;
typedef __attribute__((ext_vector_type(4)))  unsigned uint4v;
typedef __attribute__((ext_vector_type(2)))  unsigned uint2v;

union Frag16 { v16bf v; uint4v u[2]; };

#define S_WORDS 2048
#define L_CHARS 16
#define CE_DIM  128
#define CH_DIM  256
#define WE_DIM  512
#define WH_DIM  1024
#define NTAG    128
#define REC_BLOCKS 64

__device__ __forceinline__ float sigmoidf_(float x) { return 1.0f / (1.0f + __expf(-x)); }
__device__ __forceinline__ float bflo(unsigned u) { union { unsigned x; float f; } c; c.x = u << 16; return c.f; }
__device__ __forceinline__ float bfhi(unsigned u) { union { unsigned x; float f; } c; c.x = u & 0xffff0000u; return c.f; }

// ---------------- elementwise prep kernels ----------------
__global__ void k_zero_u32(unsigned* p, int n) {
    int i = blockIdx.x * blockDim.x + threadIdx.x;
    if (i < n) p[i] = 0u;
}

__global__ void k_f32_to_bf16(bf16_t* __restrict__ dst, const float* __restrict__ src, int n) {
    int i = blockIdx.x * blockDim.x + threadIdx.x;
    if (i < n) dst[i] = (bf16_t)src[i];
}

__global__ void k_add_bias(float* __restrict__ dst, const float* __restrict__ a,
                           const float* __restrict__ b, int n) {
    int i = blockIdx.x * blockDim.x + threadIdx.x;
    if (i < n) dst[i] = a[i] + b[i];
}

// cWcat[n][k] : k<128 -> c_Wih[n][k] ; else c_Whh[n][k-128]   (n in [0,1024), k in [0,384))
__global__ void k_concat_cW(bf16_t* __restrict__ dst, const float* __restrict__ wih,
                            const float* __restrict__ whh) {
    int i = blockIdx.x * blockDim.x + threadIdx.x;
    if (i >= 1024 * 384) return;
    int n = i / 384, k = i % 384;
    float v = (k < CE_DIM) ? wih[n * CE_DIM + k] : whh[n * CH_DIM + (k - CE_DIM)];
    dst[i] = (bf16_t)v;
}

// gather char embeddings -> bf16  [S, L, CE]
__global__ void k_gather_ce(bf16_t* __restrict__ dst, const int* __restrict__ char_ids,
                            const float* __restrict__ char_emb) {
    int i = blockIdx.x * blockDim.x + threadIdx.x;
    if (i >= S_WORDS * L_CHARS * CE_DIM) return;
    int w = i >> 7, e = i & (CE_DIM - 1);
    int cid = char_ids[w];
    dst[i] = (bf16_t)char_emb[cid * CE_DIM + e];
}

// wx[s][k] : k<512 -> word_emb[word_ids[s]][k] ; else last_char[s][k-512]
__global__ void k_build_wx(bf16_t* __restrict__ dst, const int* __restrict__ word_ids,
                           const float* __restrict__ word_emb, const bf16_t* __restrict__ last_bf) {
    int i = blockIdx.x * blockDim.x + threadIdx.x;
    if (i >= S_WORDS * (WE_DIM + CH_DIM)) return;
    int s = i / (WE_DIM + CH_DIM), k = i % (WE_DIM + CH_DIM);
    if (k < WE_DIM) dst[i] = (bf16_t)word_emb[(size_t)word_ids[s] * WE_DIM + k];
    else            dst[i] = last_bf[s * CH_DIM + (k - WE_DIM)];
}

// ---------------- WMMA GEMM ----------------
// C[M,N] = concat(A1[M,K1](lda1), A2[M,K2](lda2)) x Bt[N,K1+K2]^T + bias
// A, Bt bf16 row-major; C f32 row-major. Each wave computes a 32x64 tile:
// two A fragments reuse every B fragment -> 8 WMMA per 12 b128 loads per k-step.
__global__ void k_gemm_bf16(const bf16_t* __restrict__ A1, int lda1, int K1,
                            const bf16_t* __restrict__ A2, int lda2, int K2,
                            const bf16_t* __restrict__ Bt,
                            const float* __restrict__ bias,
                            float* __restrict__ C, int M, int N) {
    const int lane = threadIdx.x & 31;
    const int wave = threadIdx.x >> 5;
    const int tilesN = N >> 6;
    const int waveTile = blockIdx.x * (blockDim.x >> 5) + wave;
    const int totalTiles = (M >> 5) * tilesN;
    if (waveTile >= totalTiles) return;           // wave-uniform: EXEC stays all-ones
    const int tm = waveTile / tilesN;             // 32-row tile
    const int tn = waveTile % tilesN;             // 64-col tile
    const int K = K1 + K2;
    const int half = lane >> 4;
    const int r0 = tm * 32 + (lane & 15);
    const int r1 = r0 + 16;

    v8f acc[2][4] = {};
    for (int k = 0; k < K; k += 32) {
        // A fragments: lane holds row m=lane%16, K runs at 8*half and 16+8*half
        const bf16_t *arow0, *arow1; int kk;
        if (k < K1) { arow0 = A1 + (size_t)r0 * lda1; arow1 = A1 + (size_t)r1 * lda1; kk = k; }
        else        { arow0 = A2 + (size_t)r0 * lda2; arow1 = A2 + (size_t)r1 * lda2; kk = k - K1; }
        Frag16 fa0, fa1;
        fa0.u[0] = *(const uint4v*)(arow0 + kk + 8 * half);
        fa0.u[1] = *(const uint4v*)(arow0 + kk + 16 + 8 * half);
        fa1.u[0] = *(const uint4v*)(arow1 + kk + 8 * half);
        fa1.u[1] = *(const uint4v*)(arow1 + kk + 16 + 8 * half);
        if (k + 32 < K1) {                         // hint next A k-block into cache
            __builtin_prefetch(arow0 + kk + 32, 0, 1);
            __builtin_prefetch(arow1 + kk + 32, 0, 1);
        }
#pragma unroll
        for (int s = 0; s < 4; ++s) {
            // B fragment: lane holds col n=lane%16, K runs at 8*half and 16+8*half
            const int colB = tn * 64 + s * 16 + (lane & 15);
            const bf16_t* brow = Bt + (size_t)colB * K + k + 8 * half;
            Frag16 fb;
            fb.u[0] = *(const uint4v*)(brow);
            fb.u[1] = *(const uint4v*)(brow + 16);
            acc[0][s] = __builtin_amdgcn_wmma_f32_16x16x32_bf16(
                false, fa0.v, false, fb.v, (short)0, acc[0][s], false, false);
            acc[1][s] = __builtin_amdgcn_wmma_f32_16x16x32_bf16(
                false, fa1.v, false, fb.v, (short)0, acc[1][s], false, false);
        }
    }
#pragma unroll
    for (int mi = 0; mi < 2; ++mi) {
#pragma unroll
        for (int s = 0; s < 4; ++s) {
            const int n = tn * 64 + s * 16 + (lane & 15);
            const float b = bias ? bias[n] : 0.0f;
#pragma unroll
            for (int r = 0; r < 8; ++r) {
                const int m = tm * 32 + mi * 16 + half * 8 + r;
                C[(size_t)m * N + n] = acc[mi][s][r] + b;
            }
        }
    }
}

// ---------------- char LSTM cell (batched over 2048 words) ----------------
__global__ void k_char_cell(const float* __restrict__ gates,   // [S, 4*CH]
                            float* __restrict__ c,             // [S, CH]
                            bf16_t* __restrict__ h_bf,         // [S, CH]
                            bf16_t* __restrict__ last_bf,      // [S, CH]
                            const int* __restrict__ char_lens, int t) {
    int i = blockIdx.x * blockDim.x + threadIdx.x;
    if (i >= S_WORDS * CH_DIM) return;
    int s = i >> 8, j = i & (CH_DIM - 1);
    const float* g = gates + (size_t)s * (4 * CH_DIM);
    float ig = sigmoidf_(g[j]);
    float fg = sigmoidf_(g[CH_DIM + j]);
    float gg = tanhf(g[2 * CH_DIM + j]);
    float og = sigmoidf_(g[3 * CH_DIM + j]);
    float cv = fg * c[i] + ig * gg;
    c[i] = cv;
    float hv = og * tanhf(cv);
    h_bf[i] = (bf16_t)hv;
    if (char_lens[s] - 1 == t) last_bf[i] = (bf16_t)hv;
}

// ---------------- persistent word-LSTM recurrence ----------------
__device__ __forceinline__ void grid_barrier(unsigned* cnt, unsigned* gen, unsigned nBlocks) {
    __syncthreads();
    if (threadIdx.x == 0) {
        __threadfence();
        unsigned g = __hip_atomic_load(gen, __ATOMIC_RELAXED, __HIP_MEMORY_SCOPE_AGENT);
        unsigned prev = __hip_atomic_fetch_add(cnt, 1u, __ATOMIC_ACQ_REL, __HIP_MEMORY_SCOPE_AGENT);
        if (prev == nBlocks - 1u) {
            __hip_atomic_store(cnt, 0u, __ATOMIC_RELAXED, __HIP_MEMORY_SCOPE_AGENT);
            __hip_atomic_fetch_add(gen, 1u, __ATOMIC_RELEASE, __HIP_MEMORY_SCOPE_AGENT);
        } else {
            while (__hip_atomic_load(gen, __ATOMIC_ACQUIRE, __HIP_MEMORY_SCOPE_AGENT) == g) {
                __builtin_amdgcn_s_sleep(1);
            }
        }
    }
    __syncthreads();
}

// 64 blocks x 256 threads. Per step: rgates[4096] = h(bf16,1024) . Whh^T, then cell on 1024 elems.
__global__ void k_word_rec(const float* __restrict__ xproj,   // [S, 4*WH] (x@Wih^T + biases)
                           const bf16_t* __restrict__ Whh,    // [4*WH, WH] bf16
                           float* __restrict__ rgates,        // [4*WH]
                           float* __restrict__ c,             // [WH]
                           bf16_t* __restrict__ h_bf,         // [WH]
                           bf16_t* __restrict__ whs,          // [S, WH] bf16
                           unsigned* barCnt, unsigned* barGen) {
    __shared__ bf16_t hs[WH_DIM];
    const int tid = threadIdx.x;
    const int n = blockIdx.x * 64 + (tid >> 2);     // output gate index [0,4096)
    const int kb = (tid & 3) * 256;                 // K quarter
    const bf16_t* wrow = Whh + (size_t)n * WH_DIM + kb;

    for (int t = 0; t < S_WORDS; ++t) {
        // stage h into LDS (1024 bf16 = 2KB; 256 threads x 8B)
        ((uint2v*)hs)[tid] = ((const uint2v*)h_bf)[tid];
        __syncthreads();

        float sum = 0.0f;
        const bf16_t* hb = hs + kb;
#pragma unroll 4
        for (int k = 0; k < 256; k += 8) {
            uint4v w = *(const uint4v*)(wrow + k);
            uint4v h4 = *(const uint4v*)(hb + k);
            sum = fmaf(bflo(w[0]), bflo(h4[0]), sum);
            sum = fmaf(bfhi(w[0]), bfhi(h4[0]), sum);
            sum = fmaf(bflo(w[1]), bflo(h4[1]), sum);
            sum = fmaf(bfhi(w[1]), bfhi(h4[1]), sum);
            sum = fmaf(bflo(w[2]), bflo(h4[2]), sum);
            sum = fmaf(bfhi(w[2]), bfhi(h4[2]), sum);
            sum = fmaf(bflo(w[3]), bflo(h4[3]), sum);
            sum = fmaf(bfhi(w[3]), bfhi(h4[3]), sum);
        }
        // reduce 4 partials (lanes n*4..n*4+3 are contiguous inside a wave32)
        sum += __shfl_xor(sum, 1, 32);
        sum += __shfl_xor(sum, 2, 32);
        if ((tid & 3) == 0) rgates[n] = sum;
        grid_barrier(barCnt, barGen, REC_BLOCKS);

        // cell update: global threads 0..1023 (blocks 0..3)
        int gidx = blockIdx.x * blockDim.x + tid;
        if (gidx < WH_DIM) {
            const float* xp = xproj + (size_t)t * (4 * WH_DIM);
            float ig = sigmoidf_(xp[gidx] + rgates[gidx]);
            float fg = sigmoidf_(xp[WH_DIM + gidx] + rgates[WH_DIM + gidx]);
            float gg = tanhf(xp[2 * WH_DIM + gidx] + rgates[2 * WH_DIM + gidx]);
            float og = sigmoidf_(xp[3 * WH_DIM + gidx] + rgates[3 * WH_DIM + gidx]);
            float cv = fg * c[gidx] + ig * gg;
            c[gidx] = cv;
            float hv = og * tanhf(cv);
            h_bf[gidx] = (bf16_t)hv;
            whs[(size_t)t * WH_DIM + gidx] = (bf16_t)hv;
        }
        grid_barrier(barCnt, barGen, REC_BLOCKS);
    }
}

// ---------------- log softmax over rows of [S, NTAG] ----------------
__global__ void k_logsoftmax(const float* __restrict__ tags, float* __restrict__ out) {
    __shared__ float red[4];
    int row = blockIdx.x, tid = threadIdx.x;      // 128 threads = 4 waves
    float v = tags[row * NTAG + tid];
    float m = v;
#pragma unroll
    for (int o = 16; o > 0; o >>= 1) m = fmaxf(m, __shfl_xor(m, o, 32));
    if ((tid & 31) == 0) red[tid >> 5] = m;
    __syncthreads();
    m = fmaxf(fmaxf(red[0], red[1]), fmaxf(red[2], red[3]));
    __syncthreads();
    float e = __expf(v - m);
    float s = e;
#pragma unroll
    for (int o = 16; o > 0; o >>= 1) s += __shfl_xor(s, o, 32);
    if ((tid & 31) == 0) red[tid >> 5] = s;
    __syncthreads();
    s = red[0] + red[1] + red[2] + red[3];
    out[row * NTAG + tid] = v - m - __logf(s);
}

// ---------------- host launcher ----------------
static inline int ceil_div(int a, int b) { return (a + b - 1) / b; }

extern "C" void kernel_launch(void* const* d_in, const int* in_sizes, int n_in,
                              void* d_out, int out_size, void* d_ws, size_t ws_size,
                              hipStream_t stream) {
    (void)in_sizes; (void)n_in; (void)out_size; (void)ws_size;
    const int*   word_ids  = (const int*)d_in[0];
    const int*   char_ids  = (const int*)d_in[1];
    const int*   char_lens = (const int*)d_in[2];
    const float* char_emb  = (const float*)d_in[3];
    const float* word_emb  = (const float*)d_in[4];
    const float* c_Wih = (const float*)d_in[5];
    const float* c_Whh = (const float*)d_in[6];
    const float* c_bih = (const float*)d_in[7];
    const float* c_bhh = (const float*)d_in[8];
    const float* w_Wih = (const float*)d_in[9];
    const float* w_Whh = (const float*)d_in[10];
    const float* w_bih = (const float*)d_in[11];
    const float* w_bhh = (const float*)d_in[12];
    const float* out_W = (const float*)d_in[13];
    const float* out_b = (const float*)d_in[14];
    float* out = (float*)d_out;

    // workspace carve-up (256B aligned)
    char* base = (char*)d_ws;
    size_t off = 0;
    auto take = [&](size_t bytes) -> char* {
        char* p = base + off;
        off += (bytes + 255) & ~(size_t)255;
        return p;
    };

    // zero region: c_char | h_char(bf16) | c_rec | h_rec(bf16) | barrier cnt/gen
    const size_t ZB_CCHAR = (size_t)S_WORDS * CH_DIM * 4;   // 2 MB
    const size_t ZB_HCHAR = (size_t)S_WORDS * CH_DIM * 2;   // 1 MB
    const size_t ZB_CREC  = (size_t)WH_DIM * 4;             // 4 KB
    const size_t ZB_HREC  = (size_t)WH_DIM * 2;             // 2 KB
    const size_t ZBYTES   = ZB_CCHAR + ZB_HCHAR + ZB_CREC + ZB_HREC + 256;
    char* zbase = take(ZBYTES);
    float*    c_char  = (float*)(zbase);
    bf16_t*   h_char  = (bf16_t*)(zbase + ZB_CCHAR);
    float*    c_rec   = (float*)(zbase + ZB_CCHAR + ZB_HCHAR);
    bf16_t*   h_rec   = (bf16_t*)(zbase + ZB_CCHAR + ZB_HCHAR + ZB_CREC);
    unsigned* barCnt  = (unsigned*)(zbase + ZB_CCHAR + ZB_HCHAR + ZB_CREC + ZB_HREC);
    unsigned* barGen  = barCnt + 1;

    bf16_t* cWcat   = (bf16_t*)take((size_t)1024 * 384 * 2);
    float*  cbias   = (float*)take(1024 * 4);
    bf16_t* wWih_b  = (bf16_t*)take((size_t)4096 * 768 * 2);
    float*  wbias   = (float*)take(4096 * 4);
    bf16_t* wWhh_b  = (bf16_t*)take((size_t)4096 * 1024 * 2);
    bf16_t* outW_b  = (bf16_t*)take((size_t)NTAG * WH_DIM * 2);
    bf16_t* ce_bf   = (bf16_t*)take((size_t)S_WORDS * L_CHARS * CE_DIM * 2);
    bf16_t* last_bf = (bf16_t*)take((size_t)S_WORDS * CH_DIM * 2);
    float*  gates   = (float*)take((size_t)S_WORDS * 4 * CH_DIM * 4);
    bf16_t* wx      = (bf16_t*)take((size_t)S_WORDS * (WE_DIM + CH_DIM) * 2);
    float*  xproj   = (float*)take((size_t)S_WORDS * 4 * WH_DIM * 4);
    float*  rgates  = (float*)take(4096 * 4);
    bf16_t* whs     = (bf16_t*)take((size_t)S_WORDS * WH_DIM * 2);
    float*  tags    = (float*)take((size_t)S_WORDS * NTAG * 4);

    const int TB = 256;

    // init recurrent state + barrier counters
    {
        int n = (int)(ZBYTES / 4);
        k_zero_u32<<<ceil_div(n, TB), TB, 0, stream>>>((unsigned*)zbase, n);
    }
    // weight prep
    k_concat_cW<<<ceil_div(1024 * 384, TB), TB, 0, stream>>>(cWcat, c_Wih, c_Whh);
    k_add_bias<<<ceil_div(1024, TB), TB, 0, stream>>>(cbias, c_bih, c_bhh, 1024);
    k_f32_to_bf16<<<ceil_div(4096 * 768, TB), TB, 0, stream>>>(wWih_b, w_Wih, 4096 * 768);
    k_add_bias<<<ceil_div(4096, TB), TB, 0, stream>>>(wbias, w_bih, w_bhh, 4096);
    k_f32_to_bf16<<<ceil_div(4096 * 1024, TB), TB, 0, stream>>>(wWhh_b, w_Whh, 4096 * 1024);
    k_f32_to_bf16<<<ceil_div(NTAG * WH_DIM, TB), TB, 0, stream>>>(outW_b, out_W, NTAG * WH_DIM);
    k_gather_ce<<<ceil_div(S_WORDS * L_CHARS * CE_DIM, TB), TB, 0, stream>>>(ce_bf, char_ids, char_emb);

    // ---- char LSTM: 16 steps, each = WMMA GEMM [2048,384]x[384,1024] + cell ----
    for (int t = 0; t < L_CHARS; ++t) {
        int tiles = (S_WORDS / 32) * (1024 / 64);         // 1024 wave tiles
        k_gemm_bf16<<<tiles / 8, 256, 0, stream>>>(
            ce_bf + (size_t)t * CE_DIM, L_CHARS * CE_DIM, CE_DIM,
            h_char, CH_DIM, CH_DIM,
            cWcat, cbias, gates, S_WORDS, 4 * CH_DIM);
        k_char_cell<<<ceil_div(S_WORDS * CH_DIM, TB), TB, 0, stream>>>(
            gates, c_char, h_char, last_bf, char_lens, t);
    }

    // ---- word LSTM ----
    k_build_wx<<<ceil_div(S_WORDS * (WE_DIM + CH_DIM), TB), TB, 0, stream>>>(
        wx, word_ids, word_emb, last_bf);
    {   // input projection: [2048,768] x [768,4096]
        int tiles = (S_WORDS / 32) * (4096 / 64);         // 4096 wave tiles
        k_gemm_bf16<<<tiles / 8, 256, 0, stream>>>(
            wx, WE_DIM + CH_DIM, WE_DIM + CH_DIM,
            (const bf16_t*)nullptr, 0, 0,
            wWih_b, wbias, xproj, S_WORDS, 4 * WH_DIM);
    }
    k_word_rec<<<REC_BLOCKS, 256, 0, stream>>>(xproj, wWhh_b, rgates, c_rec, h_rec, whs,
                                               barCnt, barGen);

    // ---- output projection + log softmax ----
    {
        int tiles = (S_WORDS / 32) * (NTAG / 64);         // 128 wave tiles
        k_gemm_bf16<<<tiles / 8, 256, 0, stream>>>(
            whs, WH_DIM, WH_DIM,
            (const bf16_t*)nullptr, 0, 0,
            outW_b, out_b, tags, S_WORDS, NTAG);
    }
    k_logsoftmax<<<S_WORDS, NTAG, 0, stream>>>(tags, out);
}
